// GPT2Model_3152505995809
// MI455X (gfx1250) — compile-verified, matrix operand order checked by reading, not run on hardware
//
#include <hip/hip_runtime.h>
#include <math.h>

// ---------------------------------------------------------------------------
// Types for CDNA5 WMMA (wave32): A/B = 16 bf16 per lane, C/D = 8 f32 per lane
// ---------------------------------------------------------------------------
typedef __attribute__((ext_vector_type(16))) __bf16 bf16x16;
typedef __attribute__((ext_vector_type(8)))  float  f32x8;

#define BB    2
#define TTOK  1024
#define EE    1024
#define HH    16
#define DD    64
#define NRELC 529
#define RVOC  64
#define KMAX  32
#define INFR       1.0e10f
#define INV_SQRT_D 0.125f                 // 1/sqrt(64)
#define INV_SQRT3  0.57735026918962576f   // 1/sqrt(3)

__device__ __forceinline__ f32x8 wmma_bf16(bf16x16 a, bf16x16 b, f32x8 c) {
  return __builtin_amdgcn_wmma_f32_16x16x32_bf16(false, a, false, b, (short)0, c,
                                                 false, false);
}
__device__ __forceinline__ f32x8 zero8() {
  f32x8 v;
#pragma unroll
  for (int e = 0; e < 8; ++e) v[e] = 0.0f;
  return v;
}

// ---------------------------------------------------------------------------
// Elementwise helpers
// ---------------------------------------------------------------------------
__global__ void cast_f32_bf16_kernel(const float* __restrict__ in,
                                     __bf16* __restrict__ out, int n) {
  int i = blockIdx.x * blockDim.x + threadIdx.x;
  if (i < n) out[i] = (__bf16)in[i];
}

// rel_values [65,64] -> bf16 padded to [96,64] (pad rows zero) for K=96 WMMA
__global__ void relv_pad_kernel(const float* __restrict__ rv,
                                __bf16* __restrict__ out) {
  int i = blockIdx.x * blockDim.x + threadIdx.x;   // 96*64
  if (i >= 96 * 64) return;
  int r = i / 64;
  out[i] = (__bf16)((r < 2 * KMAX + 1) ? rv[i] : 0.0f);
}

// LayerNorm (TF style, unbiased std, /(std+eps)) -> bf16
__global__ __launch_bounds__(256)
void layernorm_kernel(const float* __restrict__ x, const float* __restrict__ w,
                      const float* __restrict__ b, __bf16* __restrict__ out) {
  __shared__ float red[8];
  int row = blockIdx.x;
  int tid = threadIdx.x, lane = tid & 31, wid = tid >> 5;
  const float* xr = x + (size_t)row * EE;

  float s = 0.f;
  for (int i = tid; i < EE; i += 256) s += xr[i];
#pragma unroll
  for (int m = 16; m >= 1; m >>= 1) s += __shfl_xor(s, m, 32);
  if (lane == 0) red[wid] = s;
  __syncthreads();
  float tot = 0.f;
#pragma unroll
  for (int i = 0; i < 8; ++i) tot += red[i];
  float u = tot / (float)EE;
  __syncthreads();

  float ss = 0.f;
  for (int i = tid; i < EE; i += 256) { float d = xr[i] - u; ss += d * d; }
#pragma unroll
  for (int m = 16; m >= 1; m >>= 1) ss += __shfl_xor(ss, m, 32);
  if (lane == 0) red[wid] = ss;
  __syncthreads();
  float sst = 0.f;
#pragma unroll
  for (int i = 0; i < 8; ++i) sst += red[i];
  float sd  = sqrtf(sst / (float)(EE - 1));
  float inv = 1.0f / (sd + 1e-6f);
  for (int i = tid; i < EE; i += 256)
    out[(size_t)row * EE + i] = (__bf16)(w[i] * (xr[i] - u) * inv + b[i]);
}

// qkv f32 [B*T, 3E] -> q,k,v bf16 [B,H,T,D]
__global__ void split_qkv_kernel(const float* __restrict__ qkv,
                                 __bf16* __restrict__ q, __bf16* __restrict__ k,
                                 __bf16* __restrict__ v) {
  int i = blockIdx.x * blockDim.x + threadIdx.x;   // B*H*T*D
  if (i >= BB * HH * TTOK * DD) return;
  int d = i & 63;
  int t = (i >> 6) & 1023;
  int h = (i >> 16) & 15;
  int b = i >> 20;
  size_t src = ((size_t)(b * TTOK + t)) * (3 * EE) + h * DD + d;
  size_t dst = (((size_t)(b * HH + h)) * TTOK + t) * DD + d;
  q[dst] = (__bf16)qkv[src];
  k[dst] = (__bf16)qkv[src + EE];
  v[dst] = (__bf16)qkv[src + 2 * EE];
}

// attnout f32 [B,H,T,D] -> y bf16 [B,T,E]
__global__ void pack_heads_kernel(const float* __restrict__ ao,
                                  __bf16* __restrict__ y) {
  int i = blockIdx.x * blockDim.x + threadIdx.x;   // B*T*E
  if (i >= BB * TTOK * EE) return;
  int col = i & 1023, row = i >> 10;
  int h = col >> 6, d = col & 63;
  int b = row >> 10, t = row & 1023;
  y[i] = (__bf16)ao[(((size_t)(b * HH + h)) * TTOK + t) * DD + d];
}

// ---------------------------------------------------------------------------
// Generic bf16 WMMA GEMM: C = A[M,K] @ B[K,N] (+bias) with epilogues.
//   Block: 256 thr = 8 waves, tile 128x64, BK=32, double-buffered LDS
//   (one barrier per K step; staging overlaps WMMA).
//   TRANSB: B stored [N,K] row-major (loaded transposed into LDS)
//   EPI: 0=store f32, 1=store bf16, 2=GELU->bf16, 3=C_f32 += val, 4=+resid->f32
// ---------------------------------------------------------------------------
template <bool TRANSB, int EPI>
__global__ __launch_bounds__(256)
void gemm_bf16_kernel(const __bf16* __restrict__ A, const __bf16* __restrict__ Bm,
                      float* Cf, __bf16* Cb,
                      const float* __restrict__ bias, const float* __restrict__ resid,
                      int M, int N, int K, int lda, int ldb, int ldc,
                      int Mstore, int Nstore,
                      long strideA, long strideB, long strideC,
                      int modA, int modB) {
  __shared__ __align__(64) __bf16 As[2][128][32];
  __shared__ __align__(64) __bf16 Bs[2][64][32];   // transposed: Bs[buf][n][k]

  int z = blockIdx.z;
  const __bf16* Ab = A  + (size_t)((modA ? (z % modA) : z)) * strideA;
  const __bf16* Bb = Bm + (size_t)((modB ? (z % modB) : z)) * strideB;

  int tid = threadIdx.x, lane = tid & 31, wid = tid >> 5;
  int lo = lane & 15, hi = lane >> 4;
  int wm = (wid >> 1) * 32, wn = (wid & 1) * 32;
  int m0 = blockIdx.y * 128, n0 = blockIdx.x * 64;

  auto stageAB = [&](int bufsel, int kk0) {
    {  // A tile (128x32)
      int row = tid >> 1, kc = (tid & 1) * 16;
      bf16x16 va;
      if (m0 + row < M) {
        va = *(const bf16x16*)(Ab + (size_t)(m0 + row) * lda + kk0 + kc);
        if (kk0 + 96 < K)
          __builtin_prefetch(Ab + (size_t)(m0 + row) * lda + kk0 + 96, 0, 0);
      } else {
#pragma unroll
        for (int e = 0; e < 16; ++e) va[e] = (__bf16)0.0f;
      }
      *(bf16x16*)(&As[bufsel][row][kc]) = va;
    }
    if (TRANSB) {  // B stored [N,K]
      int n = tid >> 2, kc = (tid & 3) * 8;
#pragma unroll
      for (int e = 0; e < 8; ++e)
        Bs[bufsel][n][kc + e] = (n0 + n < N)
                                    ? Bb[(size_t)(n0 + n) * ldb + kk0 + kc + e]
                                    : (__bf16)0.0f;
    } else {       // B stored [K,N]
      int kk = tid >> 3, nc = (tid & 7) * 8;
#pragma unroll
      for (int e = 0; e < 8; ++e) {
        int n = nc + e;
        Bs[bufsel][n][kk] = (n0 + n < N)
                                ? Bb[(size_t)(kk0 + kk) * ldb + n0 + n]
                                : (__bf16)0.0f;
      }
    }
  };

  f32x8 acc[2][2];
#pragma unroll
  for (int mi = 0; mi < 2; ++mi)
#pragma unroll
    for (int ni = 0; ni < 2; ++ni) acc[mi][ni] = zero8();

  stageAB(0, 0);
  __syncthreads();

  for (int k0 = 0; k0 < K; k0 += 32) {
    int cur = (k0 >> 5) & 1;
    if (k0 + 32 < K) stageAB(cur ^ 1, k0 + 32);

    bf16x16 af[2], bfr[2];
#pragma unroll
    for (int mi = 0; mi < 2; ++mi)
      af[mi] = *(const bf16x16*)(&As[cur][wm + mi * 16 + lo][hi * 16]);
#pragma unroll
    for (int ni = 0; ni < 2; ++ni)
      bfr[ni] = *(const bf16x16*)(&Bs[cur][wn + ni * 16 + lo][hi * 16]);
#pragma unroll
    for (int mi = 0; mi < 2; ++mi)
#pragma unroll
      for (int ni = 0; ni < 2; ++ni)
        acc[mi][ni] = wmma_bf16(af[mi], bfr[ni], acc[mi][ni]);
    __syncthreads();
  }

#pragma unroll
  for (int mi = 0; mi < 2; ++mi)
#pragma unroll
    for (int ni = 0; ni < 2; ++ni)
#pragma unroll
      for (int r = 0; r < 8; ++r) {
        int row = m0 + wm + mi * 16 + r + 8 * hi;
        int col = n0 + wn + ni * 16 + lo;
        if (row < Mstore && col < Nstore) {
          float vv = acc[mi][ni][r];
          if (bias) vv += bias[col];
          size_t off = (size_t)z * strideC + (size_t)row * ldc + col;
          if (EPI == 0) {
            Cf[off] = vv;
          } else if (EPI == 1) {
            Cb[off] = (__bf16)vv;
          } else if (EPI == 2) {
            float g = 0.5f * vv *
                      (1.0f + tanhf(0.7978845608028654f *
                                    (vv + 0.044715f * vv * vv * vv)));
            Cb[off] = (__bf16)g;
          } else if (EPI == 3) {
            Cf[off] += vv;
          } else {  // EPI == 4
            Cf[off] = vv + resid[(size_t)row * ldc + col];
          }
        }
      }
}

// ---------------------------------------------------------------------------
// Fused flash attention with tree/local-relation score terms.
//   grid (T/128, B*H), 256 thr (8 waves x 16 query rows), j-chunks of 32,
//   double-buffered K/V staging (one barrier per chunk).
//   Per chunk: S = Q Kᵀ via WMMA, score epilogue (qdotrel gather, rel_weights,
//   tds, LR gathers, causal), online softmax, LDS prelsum histogram (ds f32
//   atomics), PV via one K=32 WMMA per 16-wide d tile.
// ---------------------------------------------------------------------------
__global__ __launch_bounds__(256)
void attn_kernel(const __bf16* __restrict__ q, const __bf16* __restrict__ k,
                 const __bf16* __restrict__ v, const float* __restrict__ qdotrel,
                 const float* __restrict__ tds, const int* __restrict__ rel,
                 const int* __restrict__ lrmap, const float* __restrict__ relw,
                 const __bf16* __restrict__ lr1g, const __bf16* __restrict__ lr2g,
                 float* __restrict__ attnout, __bf16* __restrict__ prelsum) {
  __shared__ __align__(64) __bf16 Ks[2][32][64];
  __shared__ __align__(64) __bf16 Vt[2][64][32];
  __shared__ __align__(64) __bf16 Ps[8][16][32];
  __shared__ float PSums[8][16][66];
  __shared__ float RowS[8][16];

  int z = blockIdx.y;                 // b*H + h
  int b = z >> 4, h = z & 15;
  int blockBase = blockIdx.x * 128;
  int jmax = blockBase + 127;
  int tid = threadIdx.x, lane = tid & 31, wid = tid >> 5;
  int lo = lane & 15, hi = lane >> 4;
  int iw = blockBase + wid * 16;      // this wave's first query row

  for (int idx = tid; idx < 8 * 16 * 66; idx += 256) ((float*)PSums)[idx] = 0.f;

  auto stageKV = [&](int bufsel, int jbase) {
    int jl = tid >> 3, dc = (tid & 7) * 8;
    const __bf16* kp = k + ((size_t)z * TTOK + jbase + jl) * DD + dc;
    const __bf16* vp = v + ((size_t)z * TTOK + jbase + jl) * DD + dc;
#pragma unroll
    for (int e = 0; e < 8; ++e) Ks[bufsel][jl][dc + e] = kp[e];
#pragma unroll
    for (int e = 0; e < 8; ++e) Vt[bufsel][dc + e][jl] = vp[e];
    if (jbase + 64 <= jmax) __builtin_prefetch(kp + 32 * DD, 0, 0);
  };

  const __bf16* qb = q + (size_t)z * TTOK * DD;
  bf16x16 qf[2];
#pragma unroll
  for (int kh = 0; kh < 2; ++kh)
    qf[kh] = *(const bf16x16*)(qb + (size_t)(iw + lo) * DD + kh * 32 + hi * 16);

  f32x8 acc[4];
#pragma unroll
  for (int nd = 0; nd < 4; ++nd) acc[nd] = zero8();
  float mrow[8], lrow[8];
#pragma unroll
  for (int r = 0; r < 8; ++r) { mrow[r] = -INFR; lrow[r] = 0.f; }

  stageKV(0, 0);
  __syncthreads();

  for (int j0 = 0; j0 <= jmax; j0 += 32) {
    int cur = (j0 >> 5) & 1;
    if (j0 + 32 <= jmax) stageKV(cur ^ 1, j0 + 32);

    if (j0 <= iw + 15) {  // wave-uniform causal skip
      // ---- scores: two 16x16 tiles (j halves), K=64 in two WMMA steps ----
      f32x8 s[2];
      s[0] = zero8(); s[1] = zero8();
#pragma unroll
      for (int c = 0; c < 2; ++c)
#pragma unroll
        for (int kh = 0; kh < 2; ++kh) {
          bf16x16 bfrag =
              *(const bf16x16*)(&Ks[cur][c * 16 + lo][kh * 32 + hi * 16]);
          s[c] = wmma_bf16(qf[kh], bfrag, s[c]);
        }

      // ---- score epilogue ----
      float p[2][8], chunkmax[8];
#pragma unroll
      for (int r = 0; r < 8; ++r) chunkmax[r] = -3.0e38f;
#pragma unroll
      for (int c = 0; c < 2; ++c)
#pragma unroll
        for (int r = 0; r < 8; ++r) {
          int i = iw + r + 8 * hi;
          int j = j0 + c * 16 + lo;
          float val;
          if (j <= i) {
            int dj  = j - i;                      // <= 0
            int rid = (dj < -KMAX) ? 0 : (dj + KMAX);
            float w = s[c][r] + qdotrel[((size_t)z * TTOK + i) * 80 + rid];
            w *= INV_SQRT_D;
            int rv = rel[((size_t)b * TTOK + i) * TTOK + j];
            w *= relw[rv * HH + h];
            w += tds[((size_t)z * TTOK + i) * TTOK + j];
            int lm = lrmap[((size_t)b * TTOK + i) * TTOK + j];
            float l1 = (float)lr1g[((size_t)z * TTOK + i) * 576 + lm];
            float l2 = (float)lr2g[((size_t)z * 544 + lm) * TTOK + j];
            val = (w + (l1 + l2) * INV_SQRT_D) * INV_SQRT3;
          } else {
            val = -INFR;
          }
          p[c][r] = val;
          chunkmax[r] = fmaxf(chunkmax[r], val);
        }

      // ---- online softmax (row reductions across the 16-lane half) ----
      float alpha[8];
#pragma unroll
      for (int r = 0; r < 8; ++r) {
        float cm = chunkmax[r];
        cm = fmaxf(cm, __shfl_xor(cm, 1, 16));
        cm = fmaxf(cm, __shfl_xor(cm, 2, 16));
        cm = fmaxf(cm, __shfl_xor(cm, 4, 16));
        cm = fmaxf(cm, __shfl_xor(cm, 8, 16));
        float mnew = fmaxf(mrow[r], cm);
        float a = __expf(mrow[r] - mnew);
        mrow[r] = mnew;
        alpha[r] = a;
        float ps = 0.f;
#pragma unroll
        for (int c = 0; c < 2; ++c) {
          float e = __expf(p[c][r] - mnew);
          p[c][r] = e;
          ps += e;
        }
        ps += __shfl_xor(ps, 1, 16);
        ps += __shfl_xor(ps, 2, 16);
        ps += __shfl_xor(ps, 4, 16);
        ps += __shfl_xor(ps, 8, 16);
        lrow[r] = lrow[r] * a + ps;
      }
      if (lo == 0)
#pragma unroll
        for (int r = 0; r < 8; ++r) RowS[wid][hi * 8 + r] = alpha[r];
#pragma unroll
      for (int nd = 0; nd < 4; ++nd)
#pragma unroll
        for (int r = 0; r < 8; ++r) acc[nd][r] *= alpha[r];
      for (int idx = lane; idx < 16 * 66; idx += 32) {
        int rr = idx / 66;
        PSums[wid][rr][idx - rr * 66] *= RowS[wid][rr];
      }

      // ---- stash P tile + prelsum histogram ----
#pragma unroll
      for (int c = 0; c < 2; ++c)
#pragma unroll
        for (int r = 0; r < 8; ++r) {
          int rowl = r + 8 * hi, jl = c * 16 + lo;
          Ps[wid][rowl][jl] = (__bf16)p[c][r];
          int i = iw + rowl, j = j0 + jl;
          if (j <= i) {
            int dj  = j - i;
            int rid = (dj < -KMAX) ? 0 : (dj + KMAX);
            atomicAdd(&PSums[wid][rowl][rid], p[c][r]);
          }
        }

      // ---- PV: one K=32 WMMA per 16-wide d tile ----
      bf16x16 pf = *(const bf16x16*)(&Ps[wid][lo][hi * 16]);
#pragma unroll
      for (int nd = 0; nd < 4; ++nd) {
        bf16x16 vf = *(const bf16x16*)(&Vt[cur][nd * 16 + lo][hi * 16]);
        acc[nd] = wmma_bf16(pf, vf, acc[nd]);
      }
    }
    __syncthreads();
  }

  // ---- finalize: normalize, store attnout and normalized prelsum (pad 96) ----
  float invl[8];
#pragma unroll
  for (int r = 0; r < 8; ++r) invl[r] = 1.0f / lrow[r];
  if (lo == 0)
#pragma unroll
    for (int r = 0; r < 8; ++r) RowS[wid][hi * 8 + r] = invl[r];
#pragma unroll
  for (int nd = 0; nd < 4; ++nd)
#pragma unroll
    for (int r = 0; r < 8; ++r) {
      int i = iw + r + 8 * hi;
      attnout[((size_t)z * TTOK + i) * DD + nd * 16 + lo] = acc[nd][r] * invl[r];
    }
  for (int idx = lane; idx < 16 * 96; idx += 32) {
    int rr = idx / 96, cc = idx - rr * 96;
    float vv = (cc < 66) ? PSums[wid][rr][cc] * RowS[wid][rr] : 0.f;
    prelsum[((size_t)z * TTOK + iw + rr) * 96 + cc] = (__bf16)vv;
  }
}

// ---------------------------------------------------------------------------
// Host side
// ---------------------------------------------------------------------------
template <bool TB, int EPI>
static void gemm_go(const __bf16* A, const __bf16* Bm, float* Cf, __bf16* Cb,
                    const float* bias, const float* resid, int M, int N, int K,
                    int lda, int ldb, int ldc, int Mstore, int Nstore, long sA,
                    long sB, long sC, int modA, int modB, int batch,
                    hipStream_t s) {
  dim3 g((N + 63) / 64, (M + 127) / 128, batch);
  gemm_bf16_kernel<TB, EPI><<<g, 256, 0, s>>>(A, Bm, Cf, Cb, bias, resid, M, N,
                                              K, lda, ldb, ldc, Mstore, Nstore,
                                              sA, sB, sC, modA, modB);
}

static void cast_go(const float* in, __bf16* out, int n, hipStream_t s) {
  cast_f32_bf16_kernel<<<(n + 255) / 256, 256, 0, s>>>(in, out, n);
}

extern "C" void kernel_launch(void* const* d_in, const int* in_sizes, int n_in,
                              void* d_out, int out_size, void* d_ws,
                              size_t ws_size, hipStream_t stream) {
  const float* x      = (const float*)d_in[0];
  const float* tds    = (const float*)d_in[1];
  const float* LR_Q   = (const float*)d_in[2];
  const float* LR_K   = (const float*)d_in[3];
  const int*   rel    = (const int*)d_in[4];
  const int*   LR_map = (const int*)d_in[5];
  const float* Wqkv   = (const float*)d_in[6];
  const float* bqkv   = (const float*)d_in[7];
  const float* Wproj  = (const float*)d_in[8];
  const float* bproj  = (const float*)d_in[9];
  const float* relw   = (const float*)d_in[10];
  const float* relk   = (const float*)d_in[11];
  const float* relv   = (const float*)d_in[12];
  const float* ln1w   = (const float*)d_in[13];
  const float* ln1b   = (const float*)d_in[14];
  const float* ln2w   = (const float*)d_in[15];
  const float* ln2b   = (const float*)d_in[16];
  const float* Wfc    = (const float*)d_in[17];
  const float* bfc    = (const float*)d_in[18];
  const float* Wfp    = (const float*)d_in[19];
  const float* bfp    = (const float*)d_in[20];
  float* out = (float*)d_out;

  const int ROWS = BB * TTOK;        // 2048
  const int ZB   = BB * HH;          // 32
  const int MQ   = ZB * TTOK;        // 32768 (B*H*T)

  // workspace carve-up (256B aligned)
  size_t off = 0;
  char* ws = (char*)d_ws;
  auto alloc = [&](size_t bytes) -> char* {
    char* p = ws + off;
    off = (off + bytes + 255) & ~(size_t)255;
    return p;
  };
  __bf16* wqkv_b = (__bf16*)alloc((size_t)EE * 3 * EE * 2);
  __bf16* wprj_b = (__bf16*)alloc((size_t)EE * EE * 2);
  __bf16* wfc_b  = (__bf16*)alloc((size_t)EE * 4 * EE * 2);
  __bf16* wfp_b  = (__bf16*)alloc((size_t)4 * EE * EE * 2);
  __bf16* lrk_b  = (__bf16*)alloc((size_t)HH * NRELC * DD * 2);
  __bf16* lrq_b  = (__bf16*)alloc((size_t)HH * NRELC * DD * 2);
  __bf16* relk_b = (__bf16*)alloc((size_t)65 * DD * 2);
  __bf16* relv_b = (__bf16*)alloc((size_t)96 * DD * 2);
  __bf16* h_b    = (__bf16*)alloc((size_t)ROWS * EE * 2);
  float*  qkv_f  = (float*)alloc((size_t)ROWS * 3 * EE * 4);
  __bf16* q_b    = (__bf16*)alloc((size_t)MQ * DD * 2);
  __bf16* k_b    = (__bf16*)alloc((size_t)MQ * DD * 2);
  __bf16* v_b    = (__bf16*)alloc((size_t)MQ * DD * 2);
  float*  qdr_f  = (float*)alloc((size_t)MQ * 80 * 4);
  __bf16* lr1g_b = (__bf16*)alloc((size_t)ZB * TTOK * 576 * 2);
  __bf16* lr2g_b = (__bf16*)alloc((size_t)ZB * 544 * TTOK * 2);
  float*  ao_f   = (float*)alloc((size_t)MQ * DD * 4);
  __bf16* psum_b = (__bf16*)alloc((size_t)MQ * 96 * 2);
  __bf16* y_b    = (__bf16*)alloc((size_t)ROWS * EE * 2);
  float*  x1_f   = (float*)alloc((size_t)ROWS * EE * 4);
  __bf16* m_b    = (__bf16*)alloc((size_t)ROWS * 4 * EE * 2);
  (void)ws_size; (void)n_in; (void)in_sizes; (void)out_size;

  // 1) weight casts
  cast_go(Wqkv, wqkv_b, EE * 3 * EE, stream);
  cast_go(Wproj, wprj_b, EE * EE, stream);
  cast_go(Wfc, wfc_b, EE * 4 * EE, stream);
  cast_go(Wfp, wfp_b, 4 * EE * EE, stream);
  cast_go(LR_K, lrk_b, HH * NRELC * DD, stream);
  cast_go(LR_Q, lrq_b, HH * NRELC * DD, stream);
  cast_go(relk, relk_b, 65 * DD, stream);
  relv_pad_kernel<<<(96 * 64 + 255) / 256, 256, 0, stream>>>(relv, relv_b);

  // 2) LN1
  layernorm_kernel<<<ROWS, 256, 0, stream>>>(x, ln1w, ln1b, h_b);

  // 3) QKV GEMM: [2048,1024]x[1024,3072]+bias -> f32
  gemm_go<false, 0>(h_b, wqkv_b, qkv_f, nullptr, bqkv, nullptr, ROWS, 3 * EE,
                    EE, EE, 3 * EE, 3 * EE, ROWS, 3 * EE, 0, 0, 0, 0, 0, 1,
                    stream);
  split_qkv_kernel<<<(MQ * DD + 255) / 256, 256, 0, stream>>>(qkv_f, q_b, k_b,
                                                              v_b);

  // 4) qdotrel = q @ rel_keysᵀ : [32768,64]x[64,65] -> f32 (ld 80)
  gemm_go<true, 0>(q_b, relk_b, qdr_f, nullptr, nullptr, nullptr, MQ, 65, DD,
                   DD, DD, 80, MQ, 80, 0, 0, 0, 0, 0, 1, stream);

  // 5) lr1g[z] = q[z] @ LR_K[h]ᵀ : batched 32x([1024,64]x[64,529]) -> bf16
  gemm_go<true, 1>(q_b, lrk_b, nullptr, lr1g_b, nullptr, nullptr, TTOK, NRELC,
                   DD, DD, DD, 576, TTOK, 576, (long)TTOK * DD,
                   (long)NRELC * DD, (long)TTOK * 576, 0, HH, ZB, stream);

  // 6) lr2g[z] = LR_Q[h] @ k[z]ᵀ : batched 32x([529,64]x[64,1024]) -> bf16
  gemm_go<true, 1>(lrq_b, k_b, nullptr, lr2g_b, nullptr, nullptr, NRELC, TTOK,
                   DD, DD, DD, TTOK, 544, TTOK, (long)NRELC * DD,
                   (long)TTOK * DD, (long)544 * TTOK, HH, 0, ZB, stream);

  // 7) fused flash attention (scores + softmax + PV + prelsum)
  {
    dim3 g(TTOK / 128, ZB);
    attn_kernel<<<g, 256, 0, stream>>>(q_b, k_b, v_b, qdr_f, tds, rel, LR_map,
                                       relw, lr1g_b, lr2g_b, ao_f, psum_b);
  }

  // 8) attnout += prelsum @ rel_values : [32768,96]x[96,64], accumulate f32
  gemm_go<false, 3>(psum_b, relv_b, ao_f, nullptr, nullptr, nullptr, MQ, DD, 96,
                    96, DD, DD, MQ, DD, 0, 0, 0, 0, 0, 1, stream);

  // 9) merge heads -> bf16 [B,T,E]
  pack_heads_kernel<<<(ROWS * EE + 255) / 256, 256, 0, stream>>>(ao_f, y_b);

  // 10) proj + bias + residual(x) -> x1 f32
  gemm_go<false, 4>(y_b, wprj_b, x1_f, nullptr, bproj, x, ROWS, EE, EE, EE, EE,
                    EE, ROWS, EE, 0, 0, 0, 0, 0, 1, stream);

  // 11) LN2 -> h2 bf16 (reuse h_b)
  layernorm_kernel<<<ROWS, 256, 0, stream>>>(x1_f, ln2w, ln2b, h_b);

  // 12) FC + bias + GELU -> bf16
  gemm_go<false, 2>(h_b, wfc_b, nullptr, m_b, bfc, nullptr, ROWS, 4 * EE, EE,
                    EE, 4 * EE, 4 * EE, ROWS, 4 * EE, 0, 0, 0, 0, 0, 1, stream);

  // 13) FP + bias + residual(x1) -> d_out f32
  gemm_go<false, 4>(m_b, wfp_b, out, nullptr, bfp, x1_f, ROWS, EE, 4 * EE,
                    4 * EE, EE, EE, ROWS, EE, 0, 0, 0, 0, 0, 1, stream);
}
